// class_attention_22565758173226
// MI455X (gfx1250) — matrix operand order repaired and checked by reading, not compiled, not run on hardware
//
#include <hip/hip_runtime.h>

// ---------------------------------------------------------------------------
// Problem constants (match reference)
// ---------------------------------------------------------------------------
#define BB 32
#define LL 2048
#define HH 512
#define CC 50
#define CP 64               // padded class dim for bf16 att
#define ML (BB * LL)        // 65536 rows in the flattened (B*L, H) GEMMs

typedef __attribute__((ext_vector_type(16))) __bf16 v16bf;
typedef __attribute__((ext_vector_type(8)))  float  v8f;
typedef __attribute__((ext_vector_type(4)))  int    v4i;

union FragBF16 { v16bf v; unsigned int u[8]; };

__device__ __forceinline__ unsigned short f32_to_bf16(float f) {
    unsigned int u = __float_as_uint(f);
    u += 0x7FFFu + ((u >> 16) & 1u);          // round-to-nearest-even
    return (unsigned short)(u >> 16);
}
__device__ __forceinline__ float bf16_to_f32(unsigned short h) {
    return __uint_as_float(((unsigned int)h) << 16);
}

// ---------------------------------------------------------------------------
// CDNA5 async global->LDS copy (16 bytes / lane), ASYNCcnt-tracked.
// ---------------------------------------------------------------------------
typedef __attribute__((address_space(1))) v4i gbl_v4i;
typedef __attribute__((address_space(3))) v4i lds_v4i;

#if __has_builtin(__builtin_amdgcn_global_load_async_to_lds_b128)
__device__ __forceinline__ void async_copy16(const unsigned short* g, unsigned short* l) {
    gbl_v4i* gp = (gbl_v4i*)(unsigned long long)(g);
    lds_v4i* lp = (lds_v4i*)(unsigned int)(unsigned long long)(l);
    __builtin_amdgcn_global_load_async_to_lds_b128(gp, lp, 0, 0);
}
__device__ __forceinline__ void async_wait() {
#if __has_builtin(__builtin_amdgcn_s_wait_asynccnt)
    __builtin_amdgcn_s_wait_asynccnt(0);
#else
    asm volatile("s_wait_asynccnt 0x0" ::: "memory");
#endif
}
#else
__device__ __forceinline__ void async_copy16(const unsigned short* g, unsigned short* l) {
    *reinterpret_cast<uint4*>(l) = *reinterpret_cast<const uint4*>(g);
}
__device__ __forceinline__ void async_wait() {}
#endif

// ---------------------------------------------------------------------------
// Elementwise f32 -> bf16 (4 elems / thread); n must be a multiple of 4
// ---------------------------------------------------------------------------
__global__ __launch_bounds__(256)
void conv_bf16_kernel(const float* __restrict__ in, unsigned short* __restrict__ out, int n) {
    int base = (blockIdx.x * 256 + threadIdx.x) * 4;
    if (base + 3 < n) {
        float4 v = *reinterpret_cast<const float4*>(in + base);
        unsigned int lo = (unsigned int)f32_to_bf16(v.x) | ((unsigned int)f32_to_bf16(v.y) << 16);
        unsigned int hi = (unsigned int)f32_to_bf16(v.z) | ((unsigned int)f32_to_bf16(v.w) << 16);
        *reinterpret_cast<uint2*>(out + base) = make_uint2(lo, hi);
    }
}

// W3 (50 x 512) -> bf16, zero-padded to 128 x 512
__global__ __launch_bounds__(256)
void conv_pad_w3_kernel(const float* __restrict__ w3, unsigned short* __restrict__ out) {
    int i = blockIdx.x * 256 + threadIdx.x;          // 128*512 total
    int r = i >> 9, c = i & 511;
    out[i] = (r < CC) ? f32_to_bf16(w3[r * HH + c]) : (unsigned short)0;
}

// att (B,L,50) f32 -> (B,L,64) bf16, zero-padded classes
__global__ __launch_bounds__(256)
void conv_pad_att_kernel(const float* __restrict__ att, unsigned short* __restrict__ out) {
    int i = blockIdx.x * 256 + threadIdx.x;          // ML*64 total
    int l = i >> 6, c = i & 63;
    out[i] = (c < CC) ? f32_to_bf16(att[(size_t)l * CC + c]) : (unsigned short)0;
}

// ---------------------------------------------------------------------------
// Tiled WMMA GEMM, all-bf16 operands:  Out[m,n] = act( sum_k A[m,k]*W[n,k] )
//   block tile: 128 (M) x 128 (N); 8 waves as 4x2; wave tile 32 x 64
//   (2x4 frags, 8 WMMAs / 32-K step); double-buffered async LDS staging.
// ---------------------------------------------------------------------------
template<bool DO_TANH, bool RESIDUAL, bool OUT_BF16, bool GUARD_N>
__global__ __launch_bounds__(256)
void gemm_bf16_wmma(const unsigned short* __restrict__ A,     // (M, K) bf16
                    const unsigned short* __restrict__ W,     // (>=n0+128, K) bf16
                    const unsigned short* __restrict__ Res,   // bf16 residual, ld = ldRes
                    void* __restrict__ Out,
                    int K, int Nvalid, int ldOut, int ldRes) {
    __shared__ __align__(16) unsigned short Alds[2][128][32];
    __shared__ __align__(16) unsigned short Blds[2][128][32];

    const int tid  = threadIdx.x;
    const int lane = tid & 31;
    const int wave = tid >> 5;
    const int wm   = wave >> 1;        // 0..3 -> 32 rows each
    const int wn   = wave & 1;         // 0..1 -> 64 cols each
    const int half = lane >> 4;
    const int l16  = lane & 15;

    const int m0 = blockIdx.x * 128;
    const int n0 = blockIdx.y * 128;

    // staging assignment: 128 rows x 4 sixteen-byte segments = 512 copies
    const int ar0 = tid >> 2;          // rows 0..63
    const int seg = (tid & 3) * 8;     // bf16 element offset of 16B segment

    v8f acc[2][4];
    #pragma unroll
    for (int i = 0; i < 2; ++i)
        #pragma unroll
        for (int j = 0; j < 4; ++j)
            acc[i][j] = v8f{};

    auto stage = [&](int buf, int k0) {
        const unsigned short* Ab = A + (size_t)m0 * K + k0;
        const unsigned short* Wb = W + (size_t)n0 * K + k0;
        async_copy16(Ab + (size_t)ar0 * K + seg,        &Alds[buf][ar0][seg]);
        async_copy16(Ab + (size_t)(ar0 + 64) * K + seg, &Alds[buf][ar0 + 64][seg]);
        async_copy16(Wb + (size_t)ar0 * K + seg,        &Blds[buf][ar0][seg]);
        async_copy16(Wb + (size_t)(ar0 + 64) * K + seg, &Blds[buf][ar0 + 64][seg]);
    };

    const int ksteps = K >> 5;
    stage(0, 0);
    async_wait();
    __syncthreads();

    for (int kt = 0; kt < ksteps; ++kt) {
        const int cur = kt & 1;
        if (kt + 1 < ksteps) stage(1 - cur, (kt + 1) * 32);

        FragBF16 afr[2], bfr[4];
        #pragma unroll
        for (int mi = 0; mi < 2; ++mi) {
            int mrow = wm * 32 + mi * 16 + l16;
            #pragma unroll
            for (int i = 0; i < 8; ++i) {
                int kA = ((i < 4) ? (2 * i) : (16 + 2 * (i - 4))) + 8 * half;
                afr[mi].u[i] = *reinterpret_cast<const unsigned int*>(&Alds[cur][mrow][kA]);
            }
        }
        #pragma unroll
        for (int ni = 0; ni < 4; ++ni) {
            int nrow = wn * 64 + ni * 16 + l16;
            #pragma unroll
            for (int v = 0; v < 8; ++v) {
                int kB = ((v < 4) ? (2 * v) : (16 + 2 * (v - 4))) + 8 * half;
                bfr[ni].u[v] = *reinterpret_cast<const unsigned int*>(&Blds[cur][nrow][kB]);
            }
        }

        #pragma unroll
        for (int mi = 0; mi < 2; ++mi)
            #pragma unroll
            for (int ni = 0; ni < 4; ++ni)
                acc[mi][ni] = __builtin_amdgcn_wmma_f32_16x16x32_bf16(
                    false, afr[mi].v, false, bfr[ni].v,
                    (short)0, acc[mi][ni], false, false);

        async_wait();
        __syncthreads();
    }

    // epilogue (C layout: VGPR r -> row r+8*half, col l16)
    #pragma unroll
    for (int mi = 0; mi < 2; ++mi) {
        #pragma unroll
        for (int ni = 0; ni < 4; ++ni) {
            #pragma unroll
            for (int r = 0; r < 8; ++r) {
                int row = m0 + wm * 32 + mi * 16 + r + 8 * half;
                int col = n0 + wn * 64 + ni * 16 + l16;
                if (!GUARD_N || col < Nvalid) {
                    float v = acc[mi][ni][r];
                    if (DO_TANH)  v = tanhf(v);
                    if (RESIDUAL) v += bf16_to_f32(Res[(size_t)row * ldRes + col]);
                    if (OUT_BF16)
                        ((unsigned short*)Out)[(size_t)row * ldOut + col] = f32_to_bf16(v);
                    else
                        ((float*)Out)[(size_t)row * ldOut + col] = v;
                }
            }
        }
    }
}

// ---------------------------------------------------------------------------
// Fused: fill-past-len with last valid score, 3-tap avgpool along L
// (count_include_pad=False), softmax over C, mask. One thread per (b,l).
// ---------------------------------------------------------------------------
__global__ __launch_bounds__(256)
void smooth_softmax_kernel(const float* __restrict__ p3,
                           const int* __restrict__ lens,
                           float* __restrict__ att,      // (B,L,C)
                           float* __restrict__ smooth) { // (B,L,C)
    int gid = blockIdx.x * blockDim.x + threadIdx.x;
    if (gid >= BB * LL) return;
    int b = gid / LL, l = gid % LL;
    int len  = lens[b];
    int last = len - 1;

    float row[CC];
    #pragma unroll
    for (int c = 0; c < CC; ++c) row[c] = 0.0f;
    float cnt = 0.0f;
    #pragma unroll
    for (int dj = -1; dj <= 1; ++dj) {
        int j = l + dj;
        if (j < 0 || j >= LL) continue;
        int jj = (j < len) ? j : last;
        const float* src = p3 + ((size_t)b * LL + jj) * CC;
        cnt += 1.0f;
        #pragma unroll
        for (int c = 0; c < CC; ++c) row[c] += src[c];
    }
    float inv = 1.0f / cnt;
    float m = -3.402823466e38f;
    #pragma unroll
    for (int c = 0; c < CC; ++c) { row[c] *= inv; m = fmaxf(m, row[c]); }

    float s = 0.0f;
    float e[CC];
    #pragma unroll
    for (int c = 0; c < CC; ++c) { e[c] = __expf(row[c] - m); s += e[c]; }
    float scale = (l < len) ? (1.0f / s) : 0.0f;

    float* sm = smooth + (size_t)gid * CC;
    float* av = att    + (size_t)gid * CC;
    #pragma unroll
    for (int c = 0; c < CC; ++c) { sm[c] = row[c]; av[c] = e[c] * scale; }
}

// ---------------------------------------------------------------------------
// class_inputs[b,c,h] = sum_l att[b,l,c] * vals[b,l,h]   (bf16 operands)
// Block tile: 64 (C, padded) x 128 (H); 8 waves as 2x4; K = L stepped by 32.
// Operands are l-major: staging transposes through LDS with branch-free
// b32 pair loads along the contiguous dimension.
// ---------------------------------------------------------------------------
__global__ __launch_bounds__(256)
void class_inputs_wmma(const unsigned short* __restrict__ att_b,   // (B,L,CP) bf16, padded
                       const unsigned short* __restrict__ vals_b,  // (B,L,H) bf16
                       float* __restrict__ out) {                  // (B,C,H)
    __shared__ unsigned short Alds[64][32];    // [c][k]
    __shared__ unsigned short Blds[128][32];   // [h][k]

    const int tid  = threadIdx.x;
    const int lane = tid & 31;
    const int wave = tid >> 5;
    const int wm   = wave >> 2;        // 0..1 -> 32 c-rows each
    const int wn   = wave & 3;         // 0..3 -> 32 h-cols each
    const int half = lane >> 4;
    const int l16  = lane & 15;

    const int h0 = blockIdx.x * 128;
    const int b  = blockIdx.y;

    v8f acc[2][2];
    #pragma unroll
    for (int i = 0; i < 2; ++i)
        #pragma unroll
        for (int j = 0; j < 2; ++j)
            acc[i][j] = v8f{};

    for (int k0 = 0; k0 < LL; k0 += 32) {
        // A = att^T (64 x 32): 32 k-rows x 32 c-pairs = 1024 units, 4 / thread
        #pragma unroll
        for (int j = 0; j < 4; ++j) {
            int idx = j * 256 + tid;
            int k = idx >> 5, cp = (idx & 31) * 2;
            unsigned int v = *reinterpret_cast<const unsigned int*>(
                &att_b[((size_t)b * LL + k0 + k) * CP + cp]);
            Alds[cp][k]     = (unsigned short)(v & 0xFFFFu);
            Alds[cp + 1][k] = (unsigned short)(v >> 16);
        }
        // B = vals^T (128 x 32): 32 k-rows x 64 h-pairs = 2048 units, 8 / thread
        #pragma unroll
        for (int j = 0; j < 8; ++j) {
            int idx = j * 256 + tid;
            int k = idx >> 6, hp = (idx & 63) * 2;
            unsigned int v = *reinterpret_cast<const unsigned int*>(
                &vals_b[((size_t)b * LL + k0 + k) * HH + h0 + hp]);
            Blds[hp][k]     = (unsigned short)(v & 0xFFFFu);
            Blds[hp + 1][k] = (unsigned short)(v >> 16);
        }
        __syncthreads();

        FragBF16 afr[2], bfr[2];
        #pragma unroll
        for (int mi = 0; mi < 2; ++mi) {
            int mrow = wm * 32 + mi * 16 + l16;
            #pragma unroll
            for (int i = 0; i < 8; ++i) {
                int kA = ((i < 4) ? (2 * i) : (16 + 2 * (i - 4))) + 8 * half;
                afr[mi].u[i] = *reinterpret_cast<const unsigned int*>(&Alds[mrow][kA]);
            }
        }
        #pragma unroll
        for (int ni = 0; ni < 2; ++ni) {
            int nrow = wn * 32 + ni * 16 + l16;
            #pragma unroll
            for (int v = 0; v < 8; ++v) {
                int kB = ((v < 4) ? (2 * v) : (16 + 2 * (v - 4))) + 8 * half;
                bfr[ni].u[v] = *reinterpret_cast<const unsigned int*>(&Blds[nrow][kB]);
            }
        }

        #pragma unroll
        for (int mi = 0; mi < 2; ++mi)
            #pragma unroll
            for (int ni = 0; ni < 2; ++ni)
                acc[mi][ni] = __builtin_amdgcn_wmma_f32_16x16x32_bf16(
                    false, afr[mi].v, false, bfr[ni].v,
                    (short)0, acc[mi][ni], false, false);

        __syncthreads();
    }

    #pragma unroll
    for (int mi = 0; mi < 2; ++mi)
        #pragma unroll
        for (int ni = 0; ni < 2; ++ni)
            #pragma unroll
            for (int r = 0; r < 8; ++r) {
                int c = wm * 32 + mi * 16 + r + 8 * half;
                int h = h0 + wn * 32 + ni * 16 + l16;
                if (c < CC)
                    out[((size_t)b * CC + c) * HH + h] = acc[mi][ni][r];
            }
}

// ---------------------------------------------------------------------------
// Launcher
// ---------------------------------------------------------------------------
extern "C" void kernel_launch(void* const* d_in, const int* in_sizes, int n_in,
                              void* d_out, int out_size, void* d_ws, size_t ws_size,
                              hipStream_t stream) {
    (void)in_sizes; (void)n_in; (void)out_size; (void)ws_size;

    const float* keys = (const float*)d_in[0];   // (B,L,H)
    const float* vals = (const float*)d_in[1];   // (B,L,H)
    const int*   lens = (const int*)d_in[2];     // (B,)
    const float* W1   = (const float*)d_in[3];   // (H,H)
    const float* W2   = (const float*)d_in[4];   // (H,H)
    const float* W3   = (const float*)d_in[5];   // (C,H)

    // d_out layout: att (B,L,C) | class_inputs (B,C,H) | smooth (B,L,C)
    float* att    = (float*)d_out;
    float* clsout = att + (size_t)BB * LL * CC;
    float* smooth = clsout + (size_t)BB * CC * HH;

    // workspace (with buffer reuse):
    //   buf0: keys_bf16, later s      buf1: p1, later att_bf16(padded)   buf2: vals_bf16
    const size_t NHALF = (size_t)ML * HH;
    unsigned short* buf0   = (unsigned short*)d_ws;
    unsigned short* buf1   = buf0 + NHALF;
    unsigned short* vals_b = buf1 + NHALF;
    float*          p3     = (float*)(vals_b + NHALF);
    unsigned short* W1b    = (unsigned short*)(p3 + (size_t)ML * CC);
    unsigned short* W2b    = W1b + 512 * 512;
    unsigned short* W3b    = W2b + 512 * 512;      // 128 x 512, zero-padded

    dim3 blk(256);

    // 0) precision conversions (one pass, out of all hot loops)
    conv_bf16_kernel<<<(ML * HH) / 1024, blk, 0, stream>>>(keys, buf0, ML * HH);
    conv_bf16_kernel<<<(ML * HH) / 1024, blk, 0, stream>>>(vals, vals_b, ML * HH);
    conv_bf16_kernel<<<(HH * HH) / 1024, blk, 0, stream>>>(W1, W1b, HH * HH);
    conv_bf16_kernel<<<(HH * HH) / 1024, blk, 0, stream>>>(W2, W2b, HH * HH);
    conv_pad_w3_kernel<<<(128 * HH) / 256, blk, 0, stream>>>(W3, W3b);

    // 1) p1 = tanh(keys @ W1^T)
    gemm_bf16_wmma<true, false, true, false><<<dim3(ML / 128, HH / 128), blk, 0, stream>>>(
        buf0, W1b, nullptr, buf1, HH, HH, HH, HH);

    // 2) s = tanh(p1 @ W2^T) + p1   (s overwrites keys_bf16)
    gemm_bf16_wmma<true, true, true, false><<<dim3(ML / 128, HH / 128), blk, 0, stream>>>(
        buf1, W2b, buf1, buf0, HH, HH, HH, HH);

    // 3) p3 = s @ W3^T  (N padded to 128 in W3b; stores guarded to 50)
    gemm_bf16_wmma<false, false, false, true><<<dim3(ML / 128, 1), blk, 0, stream>>>(
        buf0, W3b, nullptr, p3, HH, CC, CC, CC);

    // 4) fill + avgpool + softmax + mask -> smooth, att
    smooth_softmax_kernel<<<(BB * LL) / 256, blk, 0, stream>>>(p3, lens, att, smooth);

    // 4.5) att -> bf16, padded to 64 classes (reuses p1 buffer)
    conv_pad_att_kernel<<<(ML * CP) / 256, blk, 0, stream>>>(att, buf1);

    // 5) class_inputs = att^T @ vals (per batch)
    class_inputs_wmma<<<dim3(HH / 128, BB), blk, 0, stream>>>(buf1, vals_b, clsout);
}
